// my_gat_conv_77043123356205
// MI455X (gfx1250) — compile-verified, matrix operand order checked by reading, not run on hardware
//
#include <hip/hip_runtime.h>
#include <hip/hip_bf16.h>

typedef __attribute__((ext_vector_type(16))) _Float16 v16h;
typedef __attribute__((ext_vector_type(8)))  float    v8f;

#define GAT_N    50000
#define GAT_E    800000
#define GAT_ET   (GAT_E + GAT_N)   // with self-loops
#define GAT_INS  512
#define GAT_HIDS 256
#define GAT_OUTS 128
#define NEG_SLOPE 0.2f

// ---------------- helpers ----------------

__device__ __forceinline__ unsigned f2u_ord(float f) {
    unsigned b = __float_as_uint(f);
    return (b & 0x80000000u) ? ~b : (b | 0x80000000u);
}
__device__ __forceinline__ float u2f_ord(unsigned u) {
    unsigned b = (u & 0x80000000u) ? (u & 0x7fffffffu) : ~u;
    return __uint_as_float(b);
}

// Load a 16-half WMMA fragment (A or B, both K-major / NT layout) from f32 memory.
// lane<16 : halves 0..7 = K k0+0..7,  halves 8..15 = K k0+16..23
// lane>=16: halves 0..7 = K k0+8..15, halves 8..15 = K k0+24..31
__device__ __forceinline__ v16h load_frag16(const float* __restrict__ rowp, int k0, int hi) {
    v16h f;
    const float* p0 = rowp + k0 + hi * 8;
    const float* p1 = rowp + k0 + 16 + hi * 8;
#pragma unroll
    for (int j = 0; j < 8; ++j) {
        f[j]     = (_Float16)p0[j];
        f[8 + j] = (_Float16)p1[j];
    }
    return f;
}

// ---------------- GEMM: C[M,NC] = A[M,K] * W[NC,K]^T via WMMA f16 ----------------
// grid.x = M/16 row tiles; 128 threads = 4 waves; wave w covers cols [w*NTW*16, ...)
template<int K, int NC, int NTW>
__global__ __launch_bounds__(128) void gemm_nt_wmma(const float* __restrict__ A,
                                                    const float* __restrict__ W,
                                                    float* __restrict__ C) {
    const int wave = threadIdx.x >> 5;
    const int lane = threadIdx.x & 31;
    const int ll   = lane & 15;
    const int hi   = lane >> 4;
    const int m0   = blockIdx.x * 16;
    const int n0   = wave * (NTW * 16);

    const float* arow = A + (size_t)(m0 + ll) * K;

    v8f acc[NTW];
#pragma unroll
    for (int t = 0; t < NTW; ++t) {
        v8f z = {};
        acc[t] = z;
    }

    for (int kb = 0; kb < K; kb += 32) {
        v16h af = load_frag16(arow, kb, hi);
#pragma unroll
        for (int t = 0; t < NTW; ++t) {
            const float* wrow = W + (size_t)(n0 + t * 16 + ll) * K;
            v16h bf = load_frag16(wrow, kb, hi);
            acc[t] = __builtin_amdgcn_wmma_f32_16x16x32_f16(
                false, af, false, bf, (short)0, acc[t], false, false);
        }
    }

#pragma unroll
    for (int t = 0; t < NTW; ++t) {
        const int col = n0 + t * 16 + ll;
#pragma unroll
        for (int r = 0; r < 8; ++r) {
            const int row = m0 + hi * 8 + r;
            C[(size_t)row * NC + col] = acc[t][r];
        }
    }
}

// ---------------- per-node attention logits: esrc = h.a_src, edst = h.a_dst ----------------
template<int NCc>
__global__ __launch_bounds__(256) void node_logits(const float* __restrict__ h,
                                                   const float* __restrict__ asrc,
                                                   const float* __restrict__ adst,
                                                   float* __restrict__ esrc,
                                                   float* __restrict__ edst) {
    const int gw   = (int)((blockIdx.x * blockDim.x + threadIdx.x) >> 5);
    const int lane = threadIdx.x & 31;
    if (gw >= GAT_N) return;
    const float* hr = h + (size_t)gw * NCc;
    float ss = 0.f, dd = 0.f;
#pragma unroll
    for (int j = 0; j < NCc / 32; ++j) {
        float v = hr[lane + j * 32];
        ss += v * asrc[lane + j * 32];
        dd += v * adst[lane + j * 32];
    }
#pragma unroll
    for (int off = 16; off > 0; off >>= 1) {
        ss += __shfl_xor(ss, off, 32);
        dd += __shfl_xor(dd, off, 32);
    }
    if (lane == 0) { esrc[gw] = ss; edst[gw] = dd; }
}

// ---------------- edge pass 1: leaky logits + segment max (order-preserving uint) ----------------
__global__ __launch_bounds__(256) void edge_logits_max(const int* __restrict__ ei,
                                                       const float* __restrict__ esrc,
                                                       const float* __restrict__ edst,
                                                       float* __restrict__ p,
                                                       unsigned* __restrict__ m_u) {
    const int k = blockIdx.x * blockDim.x + threadIdx.x;
    if (k >= GAT_ET) return;
    int s_, d_;
    if (k < GAT_E) { s_ = ei[k]; d_ = ei[GAT_E + k]; }
    else           { s_ = d_ = k - GAT_E; }
    float e = esrc[s_] + edst[d_];
    e = e > 0.f ? e : NEG_SLOPE * e;
    p[k] = e;
    atomicMax(&m_u[d_], f2u_ord(e));
}

// ---------------- edge pass 2: p = exp(e - m[dst]); segment sum ----------------
__global__ __launch_bounds__(256) void edge_exp_sum(const int* __restrict__ ei,
                                                    const unsigned* __restrict__ m_u,
                                                    float* __restrict__ p,
                                                    float* __restrict__ ssum) {
    const int k = blockIdx.x * blockDim.x + threadIdx.x;
    if (k >= GAT_ET) return;
    int d_;
    if (k < GAT_E) { d_ = ei[GAT_E + k]; }
    else           { d_ = k - GAT_E; }
    const float pv = __expf(p[k] - u2f_ord(m_u[d_]));
    p[k] = pv;
    atomicAdd(&ssum[d_], pv);
}

// ---------------- edge pass 3: out[dst] += (p/s[dst]) * h[src], one wave per edge ----------------
template<int NCc>
__global__ __launch_bounds__(256) void edge_aggregate(const int* __restrict__ ei,
                                                      const float* __restrict__ p,
                                                      const float* __restrict__ ssum,
                                                      const float* __restrict__ h,
                                                      float* __restrict__ out) {
    const int gw   = (int)((blockIdx.x * blockDim.x + threadIdx.x) >> 5);
    const int lane = threadIdx.x & 31;
    if (gw >= GAT_ET) return;
    int s_, d_;
    if (gw < GAT_E) { s_ = ei[gw]; d_ = ei[GAT_E + gw]; }
    else            { s_ = d_ = gw - GAT_E; }
    const float coef = p[gw] / ssum[d_];
    const float* hs = h + (size_t)s_ * NCc;
    float* od = out + (size_t)d_ * NCc;
#pragma unroll
    for (int j = 0; j < NCc / 32; ++j) {
        atomicAdd(&od[lane + j * 32], coef * hs[lane + j * 32]);
    }
}

// ---------------- bias (+ optional ReLU), in place ----------------
template<int NCc, bool RELU>
__global__ __launch_bounds__(256) void bias_act(float* __restrict__ x,
                                                const float* __restrict__ b,
                                                size_t total) {
    const size_t i = (size_t)blockIdx.x * blockDim.x + threadIdx.x;
    if (i >= total) return;
    float v = x[i] + b[i % NCc];
    if (RELU) v = v > 0.f ? v : 0.f;
    x[i] = v;
}

// ---------------- host-side orchestration ----------------

extern "C" void kernel_launch(void* const* d_in, const int* in_sizes, int n_in,
                              void* d_out, int out_size, void* d_ws, size_t ws_size,
                              hipStream_t stream) {
    (void)in_sizes; (void)n_in; (void)out_size; (void)ws_size;

    const float* x      = (const float*)d_in[0];
    const int*   ei     = (const int*)d_in[1];
    // d_in[2] = edge_attr (unused: edge_dim=None in reference)
    const float* W1     = (const float*)d_in[3];
    const float* a1_src = (const float*)d_in[4];
    const float* a1_dst = (const float*)d_in[5];
    const float* b1     = (const float*)d_in[6];
    const float* W2     = (const float*)d_in[7];
    const float* a2_src = (const float*)d_in[8];
    const float* a2_dst = (const float*)d_in[9];
    const float* b2     = (const float*)d_in[10];
    float* out = (float*)d_out;

    // workspace carving (256B aligned)
    char* ws = (char*)d_ws;
    size_t off = 0;
    auto carve = [&](size_t bytes) -> char* {
        char* p = ws + off;
        off += (bytes + 255) & ~(size_t)255;
        return p;
    };
    float*    h1   = (float*)carve((size_t)GAT_N * GAT_HIDS * 4);
    float*    agg1 = (float*)carve((size_t)GAT_N * GAT_HIDS * 4);  // becomes layer-2 activations
    float*    h2   = (float*)carve((size_t)GAT_N * GAT_OUTS * 4);
    float*    esrc = (float*)carve((size_t)GAT_N * 4);
    float*    edst = (float*)carve((size_t)GAT_N * 4);
    unsigned* m_u  = (unsigned*)carve((size_t)GAT_N * 4);
    float*    ssum = (float*)carve((size_t)GAT_N * 4);
    float*    pbuf = (float*)carve((size_t)GAT_ET * 4);

    const int edge_blk   = (GAT_ET + 255) / 256;           // 1 thread / edge
    const int edgew_blk  = (GAT_ET * 32 + 255) / 256;      // 1 wave / edge
    const int nodew_blk  = (GAT_N * 32 + 255) / 256;       // 1 wave / node

    // ===================== Layer 1 =====================
    hipMemsetAsync(agg1, 0, (size_t)GAT_N * GAT_HIDS * 4, stream);
    hipMemsetAsync(m_u,  0, (size_t)GAT_N * 4, stream);    // ord-uint 0 == -NaN (minimum)
    hipMemsetAsync(ssum, 0, (size_t)GAT_N * 4, stream);

    gemm_nt_wmma<GAT_INS, GAT_HIDS, 4><<<GAT_N / 16, 128, 0, stream>>>(x, W1, h1);
    node_logits<GAT_HIDS><<<nodew_blk, 256, 0, stream>>>(h1, a1_src, a1_dst, esrc, edst);
    edge_logits_max<<<edge_blk, 256, 0, stream>>>(ei, esrc, edst, pbuf, m_u);
    edge_exp_sum<<<edge_blk, 256, 0, stream>>>(ei, m_u, pbuf, ssum);
    edge_aggregate<GAT_HIDS><<<edgew_blk, 256, 0, stream>>>(ei, pbuf, ssum, h1, agg1);
    bias_act<GAT_HIDS, true><<<(int)(((size_t)GAT_N * GAT_HIDS + 255) / 256), 256, 0, stream>>>(
        agg1, b1, (size_t)GAT_N * GAT_HIDS);

    // ===================== Layer 2 =====================
    hipMemsetAsync(out,  0, (size_t)GAT_N * GAT_OUTS * 4, stream);
    hipMemsetAsync(m_u,  0, (size_t)GAT_N * 4, stream);
    hipMemsetAsync(ssum, 0, (size_t)GAT_N * 4, stream);

    gemm_nt_wmma<GAT_HIDS, GAT_OUTS, 2><<<GAT_N / 16, 128, 0, stream>>>(agg1, W2, h2);
    node_logits<GAT_OUTS><<<nodew_blk, 256, 0, stream>>>(h2, a2_src, a2_dst, esrc, edst);
    edge_logits_max<<<edge_blk, 256, 0, stream>>>(ei, esrc, edst, pbuf, m_u);
    edge_exp_sum<<<edge_blk, 256, 0, stream>>>(ei, m_u, pbuf, ssum);
    edge_aggregate<GAT_OUTS><<<edgew_blk, 256, 0, stream>>>(ei, pbuf, ssum, h2, out);
    bias_act<GAT_OUTS, false><<<(int)(((size_t)GAT_N * GAT_OUTS + 255) / 256), 256, 0, stream>>>(
        out, b2, (size_t)GAT_N * GAT_OUTS);
}
